// GlobalPointer_63728724738699
// MI455X (gfx1250) — compile-verified, matrix operand order checked by reading, not run on hardware
//
#include <hip/hip_runtime.h>

// ---------------------------------------------------------------------------
// GlobalPointer attention scorer for MI455X (gfx1250, wave32, WMMA).
//   out = mask( rope(hidden @ W^T + b).q  @  rope(...).k^T )
// Phases:
//   1) f32->f16 convert of hidden and W        (bandwidth trivial, feeds WMMA)
//   2) projection GEMM 8192x1152x768 via v_wmma_f32_16x16x32_f16
//   3) RoPE rotate + scatter to q/k [B,T,S,D] f16
//   4) 72x batched 1024x1024x64 logits GEMM via WMMA + causal/pad mask,
//      lower-triangular tiles short-circuited to NEG (no compute).
// ---------------------------------------------------------------------------

typedef __attribute__((ext_vector_type(8)))  _Float16 v8h;
typedef __attribute__((ext_vector_type(16))) _Float16 v16h;
typedef __attribute__((ext_vector_type(8)))  float    v8f;

constexpr int Bc = 8;
constexpr int Sc = 1024;
constexpr int Hc = 768;
constexpr int Tc = 9;
constexpr int Dc = 64;
constexpr int N2 = 2 * Tc * Dc;          // 1152 projection columns
constexpr float NEGV = -1e12f;

// Load a 16x32 f16 fragment in the CDNA5 WMMA A-layout (ISA 7.12.2):
// lane<16 : row = lane,   K chunks {0..7} and {16..23}
// lane>=16: row = lane-16,K chunks {8..15} and {24..31}
// Same loader serves the B operand when the source is stored N x K (i.e.
// column-major B), which holds for both GEMMs here.
__device__ __forceinline__ v16h load_frag_rowK(const _Float16* __restrict__ base,
                                               int ldk, int lane) {
  const int row = lane & 15;
  const int hi  = lane >> 4;
  const _Float16* p = base + (size_t)row * ldk + hi * 8;
  union { v16h v; v8h h[2]; } u;
  u.h[0] = *(const v8h*)(p);        // 16B load
  u.h[1] = *(const v8h*)(p + 16);   // 16B load
  return u.v;
}

// ---------------- Phase 1: f32 -> f16 convert (hidden, W) ------------------
__global__ void k_convert(const float* __restrict__ hidden,
                          const float* __restrict__ W,
                          _Float16* __restrict__ hidden_h,
                          _Float16* __restrict__ W_h) {
  const int nh = Bc * Sc * Hc;          // 6,291,456
  const int nw = N2 * Hc;               //   884,736
  for (int i = blockIdx.x * blockDim.x + threadIdx.x; i < nh + nw;
       i += gridDim.x * blockDim.x) {
    if (i < nh) hidden_h[i] = (_Float16)hidden[i];
    else        W_h[i - nh] = (_Float16)W[i - nh];
  }
}

// ---------------- Phase 2: projection GEMM Y = Ah @ Wh^T + bias ------------
// One wave computes a 16(M) x 64(N) strip: A fragment reused by 4 WMMAs.
__global__ void __launch_bounds__(32)
k_proj(const _Float16* __restrict__ Ah,     // [8192, 768]
       const _Float16* __restrict__ Wh,     // [1152, 768]  (N x K)
       const float*    __restrict__ bias,   // [1152]
       float*          __restrict__ Y) {    // [8192, 1152]
  const int lane = threadIdx.x & 31;
  const int mt   = blockIdx.x;              // 0..511  (M tiles)
  const int nt0  = blockIdx.y * 4;          // 0..68   (N tiles, groups of 4)

  v8f acc[4] = {};
  const _Float16* arow = Ah + (size_t)mt * 16 * Hc;

  for (int k0 = 0; k0 < Hc; k0 += 32) {
    v16h a = load_frag_rowK(arow + k0, Hc, lane);
#pragma unroll
    for (int j = 0; j < 4; ++j) {
      v16h bfr = load_frag_rowK(Wh + (size_t)(nt0 + j) * 16 * Hc + k0, Hc, lane);
      acc[j] = __builtin_amdgcn_wmma_f32_16x16x32_f16(
          false, a, false, bfr, (short)0, acc[j], false, false);
    }
  }

  const int col = lane & 15;
  const int hi  = lane >> 4;
#pragma unroll
  for (int j = 0; j < 4; ++j) {
    const int n  = (nt0 + j) * 16 + col;
    const float bn = bias[n];
#pragma unroll
    for (int v = 0; v < 8; ++v) {
      const int m = mt * 16 + v + 8 * hi;   // f32 C/D layout: VGPR v -> row
      Y[(size_t)m * N2 + n] = acc[j][v] + bn;
    }
  }
}

// ---------------- Phase 3: RoPE rotate + layout to q/k [B,T,S,D] f16 -------
__global__ void k_rope(const float* __restrict__ Y,      // [B*S, 1152]
                       _Float16* __restrict__ qh,        // [B,T,S,D]
                       _Float16* __restrict__ kh) {      // [B,T,S,D]
  const int total = Bc * Tc * Sc * Dc;                   // 4,718,592
  for (int i = blockIdx.x * blockDim.x + threadIdx.x; i < total;
       i += gridDim.x * blockDim.x) {
    const int d = i & (Dc - 1);
    const int s = (i >> 6) & (Sc - 1);
    const int t = (i >> 16) % Tc;
    const int b = i / (Tc * Sc * Dc);

    const float* yrow = Y + ((size_t)(b * Sc + s)) * N2 + t * 2 * Dc;
    const float qv = yrow[2 * d];
    const float kv = yrow[2 * d + 1];

    // rot(x)[d] = x[d]*cos + x2[d]*sin ; x2 = cat([-x[1::2], x[::2]])
    int dp; float sgn;
    if (d < Dc / 2) { dp = 2 * d + 1;        sgn = -1.f; }
    else            { dp = 2 * (d - Dc / 2); sgn =  1.f; }
    const float q2 = sgn * yrow[2 * dp];
    const float k2 = sgn * yrow[2 * dp + 1];

    const float theta = powf(10000.f, -(float)(2 * (d >> 1)) / (float)Dc);
    const float ang = (float)s * theta;
    float sn, cs;
    sincosf(ang, &sn, &cs);

    qh[i] = (_Float16)(qv * cs + q2 * sn);
    kh[i] = (_Float16)(kv * cs + k2 * sn);
  }
}

// ---------------- Phase 4: logits = q @ k^T, causal + pad mask -------------
// 8 waves / block, one 16x16 tile per wave. Tiles strictly below the
// diagonal skip all loads/WMMA and just store NEG.
__global__ void __launch_bounds__(256)
k_logits(const _Float16* __restrict__ qh,   // [B*T, S, D]
         const _Float16* __restrict__ kh,   // [B*T, S, D]
         const int*      __restrict__ am,   // [B, S]
         float*          __restrict__ out) {// [B*T, S, S]
  const int lane = threadIdx.x & 31;
  const int wave = threadIdx.x >> 5;
  const int nt = blockIdx.x * 8 + wave;     // 0..63
  const int mt = blockIdx.y;                // 0..63
  const int bt = blockIdx.z;                // 0..71
  const int b  = bt / Tc;

  const int col = lane & 15;
  const int hi  = lane >> 4;
  const int n   = nt * 16 + col;
  float* obase = out + ((size_t)bt * Sc + (size_t)mt * 16) * Sc;

  if (mt > nt) {                            // fully masked tile: no compute
#pragma unroll
    for (int v = 0; v < 8; ++v)
      obase[(size_t)(v + 8 * hi) * Sc + n] = NEGV;
    return;
  }

  const _Float16* qbase = qh + ((size_t)bt * Sc + (size_t)mt * 16) * Dc;
  const _Float16* kbase = kh + ((size_t)bt * Sc + (size_t)nt * 16) * Dc;

  v8f acc = {};
#pragma unroll
  for (int k0 = 0; k0 < Dc; k0 += 32) {
    v16h a  = load_frag_rowK(qbase + k0, Dc, lane);
    v16h bf = load_frag_rowK(kbase + k0, Dc, lane);
    acc = __builtin_amdgcn_wmma_f32_16x16x32_f16(
        false, a, false, bf, (short)0, acc, false, false);
  }

  const int amn = am[b * Sc + n];
#pragma unroll
  for (int v = 0; v < 8; ++v) {
    const int m = mt * 16 + v + 8 * hi;
    const bool keep = (m <= n) && amn && am[b * Sc + m];
    obase[(size_t)(v + 8 * hi) * Sc + n] = keep ? acc[v] : NEGV;
  }
}

// ---------------------------------------------------------------------------
extern "C" void kernel_launch(void* const* d_in, const int* in_sizes, int n_in,
                              void* d_out, int out_size, void* d_ws, size_t ws_size,
                              hipStream_t stream) {
  const float* hidden = (const float*)d_in[0];   // [B,S,H]
  const int*   amask  = (const int*)d_in[1];     // [B,S]
  const float* W      = (const float*)d_in[2];   // [2*T*D, H]
  const float* bias   = (const float*)d_in[3];   // [2*T*D]
  float* out = (float*)d_out;                    // [B,T,S,S]

  char* ws = (char*)d_ws;
  size_t off = 0;
  auto take = [&](size_t bytes) -> char* {
    char* p = ws + off;
    off += (bytes + 255) & ~(size_t)255;
    return p;
  };
  _Float16* hidden_h = (_Float16*)take((size_t)Bc * Sc * Hc * 2);   // 12.6 MB
  _Float16* W_h      = (_Float16*)take((size_t)N2 * Hc * 2);        //  1.8 MB
  float*    Y        = (float*)   take((size_t)Bc * Sc * N2 * 4);   // 37.7 MB
  _Float16* qh       = (_Float16*)take((size_t)Bc * Tc * Sc * Dc * 2); // 9.4 MB
  _Float16* kh       = (_Float16*)take((size_t)Bc * Tc * Sc * Dc * 2); // 9.4 MB
  (void)ws_size; (void)in_sizes; (void)n_in; (void)out_size;

  // Phase 1: convert to f16
  {
    const int ntot = Bc * Sc * Hc + N2 * Hc;
    k_convert<<<(ntot + 255) / 256, 256, 0, stream>>>(hidden, W, hidden_h, W_h);
  }
  // Phase 2: projection GEMM (WMMA)
  {
    dim3 grid(Bc * Sc / 16, N2 / 64);   // (512, 18)
    k_proj<<<grid, 32, 0, stream>>>(hidden_h, W_h, bias, Y);
  }
  // Phase 3: RoPE
  {
    const int ntot = Bc * Tc * Sc * Dc;
    k_rope<<<(ntot + 255) / 256, 256, 0, stream>>>(Y, qh, kh);
  }
  // Phase 4: logits GEMM + mask (WMMA)
  {
    dim3 grid(Sc / 16 / 8, Sc / 16, Bc * Tc);   // (8, 64, 72)
    k_logits<<<grid, 256, 0, stream>>>(qh, kh, amask, out);
  }
}
